// GATLayer_61091614818555
// MI455X (gfx1250) — matrix operand order repaired and checked by reading, not compiled, not run on hardware
//
#include <hip/hip_runtime.h>

// GAT layer, fused flash-style for MI455X (gfx1250, wave32).
// N=4096, D=512, H=8, F=64.

#define ALPHA_    0.2f
#define NEG_FILL_ -9.0e15f
#define N_  4096
#define D_  512
#define H_  8
#define F_  64

typedef __attribute__((ext_vector_type(16))) __bf16 v16bf;
typedef __attribute__((ext_vector_type(8)))  __bf16 v8bf;
typedef __attribute__((ext_vector_type(8)))  float  v8f;
typedef __attribute__((ext_vector_type(4)))  int    v4i;
typedef __attribute__((ext_vector_type(8)))  int    v8i;
typedef __attribute__((ext_vector_type(4)))  unsigned int v4u;

// ---------------- Kernel 0a: h (f32) -> bf16 row-major ----------------
__global__ __launch_bounds__(256) void cvt_h_kernel(const float* __restrict__ h,
                                                    __bf16* __restrict__ hb) {
    int idx = blockIdx.x * 256 + threadIdx.x;   // N_*D_ threads
    hb[idx] = (__bf16)h[idx];
}

// ---------------- Kernel 0b: W[h][d][f] -> Wt[h][f][d] bf16 ----------------
__global__ __launch_bounds__(256) void cvt_w_kernel(const float* __restrict__ W,
                                                    __bf16* __restrict__ Wt) {
    int idx = blockIdx.x * 256 + threadIdx.x;   // H_*F_*D_ threads
    int hd  = idx / (F_ * D_);
    int rem = idx - hd * (F_ * D_);
    int f   = rem / D_;
    int d   = rem - f * D_;
    Wt[idx] = (__bf16)W[(hd * D_ + d) * F_ + f];
}

// ---------------- Kernel 1: hp = h @ W[h]  (+ f_src/f_dst) ----------------
// One wave per (16-row tile, head). Output hpT[h][f][n] bf16 (K-major for k2's B).
__global__ __launch_bounds__(32) void gemm1_kernel(const __bf16* __restrict__ hb,
                                                   const __bf16* __restrict__ Wt,
                                                   const float*  __restrict__ a,
                                                   __bf16* __restrict__ hpT,
                                                   float* __restrict__ fsrc,
                                                   float* __restrict__ fdst) {
    const int n0   = blockIdx.x * 16;
    const int head = blockIdx.y;
    const int lane = threadIdx.x;
    const int lm   = lane & 15;
    const int half = lane >> 4;

    v8f acc[4] = {v8f{}, v8f{}, v8f{}, v8f{}};

    // A operand: 16x32 bf16, lane=M=lm, K elems {0..7,16..23}+half*8 (ISA 7.12.2)
    const __bf16* arow  = hb + (size_t)(n0 + lm) * D_ + half * 8;
    // B operand: 32x16 bf16, lane=N(col f)=lm, K elems {0..7,16..23}+half*8
    const __bf16* bbase = Wt + ((size_t)head * F_ + lm) * D_ + half * 8;

    for (int d0 = 0; d0 < D_; d0 += 32) {
        // issue all loads for this K-step first, then the 4 WMMAs
        v8bf a0 = *(const v8bf*)(arow + d0);        // K = d0+half*8   + 0..7
        v8bf a1 = *(const v8bf*)(arow + d0 + 16);   // K = d0+16+half*8+ 0..7
        v16bf Bv[4];
#pragma unroll
        for (int b = 0; b < 4; ++b) {
            v8bf b0 = *(const v8bf*)(bbase + (size_t)b * 16 * D_ + d0);
            v8bf b1 = *(const v8bf*)(bbase + (size_t)b * 16 * D_ + d0 + 16);
            for (int t = 0; t < 8; ++t) { Bv[b][t] = b0[t]; Bv[b][t + 8] = b1[t]; }
        }
        v16bf A;
#pragma unroll
        for (int t = 0; t < 8; ++t) { A[t] = a0[t]; A[t + 8] = a1[t]; }
#pragma unroll
        for (int b = 0; b < 4; ++b)
            acc[b] = __builtin_amdgcn_wmma_f32_16x16x32_bf16(
                false, A, false, Bv[b], (short)0, acc[b], false, false);
    }

    // C layout: reg r, lanes0-15: M=r; lanes16-31: M=r+8; N(col)=lm (ISA 7.12.2)
    __shared__ float sm[16 * 64];
#pragma unroll
    for (int b = 0; b < 4; ++b)
#pragma unroll
        for (int r = 0; r < 8; ++r) {
            int m = r + half * 8;
            int f = b * 16 + lm;
            float v = acc[b][r];
            sm[m * 64 + f] = v;
            hpT[((size_t)head * F_ + f) * N_ + n0 + m] = (__bf16)v;
        }
    __syncthreads();

    if (lane < 16) {                  // one row per lane: dot with a_src/a_dst
        float fs = 0.f, fd = 0.f;
        const float* as = a + head * 2 * F_;
#pragma unroll
        for (int f = 0; f < F_; ++f) {
            float v = sm[lane * 64 + f];
            fs += v * as[f];
            fd += v * as[F_ + f];
        }
        fsrc[head * N_ + n0 + lane] = fs;
        fdst[head * N_ + n0 + lane] = fd;
    }
}

// ---------------- Kernel 2: fused masked softmax + att@hp + ELU ----------------
// 256 threads = 8 waves; wave w = head w; block = one 16-row tile.
// adj tile DMA'd into LDS by the Tensor Data Mover (double-buffered), shared by
// all 8 heads (adj is head-invariant).
__global__ __launch_bounds__(256) void attn_kernel(const int*    __restrict__ adj,
                                                   const float*  __restrict__ fsrc,
                                                   const float*  __restrict__ fdst,
                                                   const __bf16* __restrict__ hpT,
                                                   float* __restrict__ out) {
    const int i0   = blockIdx.x * 16;
    const int tid  = threadIdx.x;
    const int head = tid >> 5;
    const int lane = tid & 31;
    const int lm   = lane & 15;
    const int half = lane >> 4;

    __shared__ __align__(16) int adjS[2][16 * 128];   // double-buffered TDM target

    const float fs_i = fsrc[head * N_ + i0 + lm];   // this lane's row logit
    const float* fd  = fdst + head * N_;

    float m = -3.0e38f;   // running row max
    float l = 0.f;        // running row sum
    v8f O[4] = {v8f{}, v8f{}, v8f{}, v8f{}};

    // ---- TDM 2D tile descriptor (D#), CDNA5 ISA ----
    // group1: data_size=4B, tensor_dim0=dim1=4096, tile=128x16, dim0_stride=4096
    const v8i g1  = {0x20000, 0x10000000, 0x10000000, 0x00800000, 16, N_, 0, 0};
    const v4i gz4 = {0, 0, 0, 0};
    const v8i gz8 = {0, 0, 0, 0, 0, 0, 0, 0};

    const int ntiles = N_ / 128;

    if (head == 0) {   // wave 0 drives the DMA (TDM ignores EXEC; uniform branch)
        unsigned long long ga =
            (unsigned long long)(uintptr_t)(adj + (size_t)i0 * N_);
        unsigned int lds0 = (unsigned int)(uintptr_t)&adjS[0][0];
        v4u g0 = {1u, lds0, (unsigned int)ga,
                  (unsigned int)((ga >> 32) & 0x1FFFFFFu) | (2u << 30)};
        __builtin_amdgcn_tensor_load_to_lds(g0, g1, gz4, gz4, gz8, 0);   // tile 0
    }

    for (int t = 0; t < ntiles; ++t) {
        const int jb  = t * 128;
        const int buf = t & 1;

        if (head == 0) {
            if (t + 1 < ntiles) {   // kick DMA for tile t+1 into the other buffer
                unsigned long long ga = (unsigned long long)(uintptr_t)
                    (adj + (size_t)i0 * N_ + (jb + 128));
                unsigned int lds1 = (unsigned int)(uintptr_t)&adjS[buf ^ 1][0];
                v4u g0 = {1u, lds1, (unsigned int)ga,
                          (unsigned int)((ga >> 32) & 0x1FFFFFFu) | (2u << 30)};
                __builtin_amdgcn_tensor_load_to_lds(g0, g1, gz4, gz4, gz8, 0);
                __builtin_amdgcn_s_wait_tensorcnt(1);   // tile t landed
            } else {
                __builtin_amdgcn_s_wait_tensorcnt(0);
            }
        }
        __syncthreads();                       // publish tile t to all 8 waves
        const int* adjT = &adjS[buf][0];

#pragma unroll
        for (int sb = 0; sb < 4; ++sb) {
            const int j0 = jb + sb * 32;

            // V operand from hpT[h][f][j]: issue loads early, hide under exp VALU
            const __bf16* vb = hpT + ((size_t)head * F_ + lm) * N_ + j0 + half * 8;
            v16bf Bv[4];
#pragma unroll
            for (int b = 0; b < 4; ++b) {
                v8bf b0 = *(const v8bf*)(vb + (size_t)b * 16 * N_);
                v8bf b1 = *(const v8bf*)(vb + (size_t)b * 16 * N_ + 16);
                for (int t = 0; t < 8; ++t) { Bv[b][t] = b0[t]; Bv[b][t + 8] = b1[t]; }
            }

            // e-values in WMMA-A element order: t -> K = (t<8?t:t+8)+half*8
            float ev[16];
            float lmax = -3.0e38f;
#pragma unroll
            for (int u = 0; u < 16; ++u) {
                int k  = ((u < 8) ? u : u + 8) + half * 8;
                int jl = sb * 32 + k;
                float e = fs_i + fd[jb + jl];
                e = (e > 0.f) ? e : ALPHA_ * e;                 // leaky_relu
                e = (adjT[lm * 128 + jl] > 0) ? e : NEG_FILL_;  // mask
                ev[u] = e;
                lmax = fmaxf(lmax, e);
            }
            // combine the two K-halves of this row (wave32 xor-16)
            lmax = fmaxf(lmax, __shfl_xor(lmax, 16, 32));
            float mn = fmaxf(m, lmax);

            v16bf P;
            float lsum = 0.f;
#pragma unroll
            for (int u = 0; u < 16; ++u) {
                float p = __expf(ev[u] - mn);
                lsum += p;
                P[u] = (__bf16)p;
            }
            lsum += __shfl_xor(lsum, 16, 32);

            float scale = __expf(m - mn);
            l = l * scale + lsum;
            m = mn;

            // rescale O: row of O[b][r] is r+half*8; scale lives at lane==row
#pragma unroll
            for (int r = 0; r < 8; ++r) {
                float sr = __shfl(scale, r + half * 8, 32);
                O[0][r] *= sr; O[1][r] *= sr; O[2][r] *= sr; O[3][r] *= sr;
            }

#pragma unroll
            for (int b = 0; b < 4; ++b)
                O[b] = __builtin_amdgcn_wmma_f32_16x16x32_bf16(
                    false, P, false, Bv[b], (short)0, O[b], false, false);
        }
        __syncthreads();   // all waves done with buf before TDM may overwrite it
    }

    // finalize: divide by l (per row), ELU, store out[n][h*F+f] (f32)
    float linv = 1.0f / l;
#pragma unroll
    for (int r = 0; r < 8; ++r) {
        float li = __shfl(linv, r + half * 8, 32);
        int row  = i0 + r + half * 8;
#pragma unroll
        for (int b = 0; b < 4; ++b) {
            float v = O[b][r] * li;
            v = (v > 0.f) ? v : (__expf(v) - 1.0f);   // elu(alpha=1)
            out[(size_t)row * (H_ * F_) + head * F_ + b * 16 + lm] = v;
        }
    }
}

// ---------------- launcher ----------------
extern "C" void kernel_launch(void* const* d_in, const int* in_sizes, int n_in,
                              void* d_out, int out_size, void* d_ws, size_t ws_size,
                              hipStream_t stream) {
    const float* h   = (const float*)d_in[0];   // [N,D] f32
    const int*   adj = (const int*)  d_in[1];   // [N,N] i32
    const float* W   = (const float*)d_in[2];   // [H,D,F] f32
    const float* a   = (const float*)d_in[3];   // [H,2F] f32
    float* out = (float*)d_out;                 // [N, H*F] f32

    // workspace partition
    __bf16* hb   = (__bf16*)d_ws;               // N*D bf16      (4 MB)
    __bf16* Wt   = hb  + (size_t)N_ * D_;       // H*F*D bf16    (0.5 MB)
    __bf16* hpT  = Wt  + (size_t)H_ * F_ * D_;  // H*F*N bf16    (4 MB)
    float*  fsrc = (float*)(hpT + (size_t)H_ * F_ * N_);
    float*  fdst = fsrc + (size_t)H_ * N_;

    cvt_h_kernel<<<(N_ * D_) / 256, 256, 0, stream>>>(h, hb);
    cvt_w_kernel<<<(H_ * F_ * D_) / 256, 256, 0, stream>>>(W, Wt);
    gemm1_kernel<<<dim3(N_ / 16, H_), 32, 0, stream>>>(hb, Wt, a, hpT, fsrc, fdst);
    attn_kernel<<<N_ / 16, 256, 0, stream>>>(adj, fsrc, fdst, hpT, out);
}
